// GCN_70824010711505
// MI455X (gfx1250) — matrix operand order repaired and checked by reading, not compiled, not run on hardware
//
#include <hip/hip_runtime.h>
#include <hip/hip_bf16.h>

#define HDIM 64
#define F_IN 16
#define N_GRAPHS 512
#define BN_EPS 1e-5f

typedef __attribute__((ext_vector_type(16))) _Float16 v16h;
typedef __attribute__((ext_vector_type(8)))  _Float16 v8h;
typedef __attribute__((ext_vector_type(8)))  float    v8f;

// ---------- helpers: order-preserving float<->uint for atomicMax pooling ----------
__device__ __forceinline__ unsigned f2ord(float f) {
  unsigned u = __float_as_uint(f);
  return (u & 0x80000000u) ? ~u : (u | 0x80000000u);
}
__device__ __forceinline__ float ord2f(unsigned u) {
  unsigned b = (u & 0x80000000u) ? (u ^ 0x80000000u) : ~u;
  return __uint_as_float(b);
}

// ---------- fills ----------
__global__ void fill_f32(float* p, float v, long long n) {
  long long t = blockIdx.x * (long long)blockDim.x + threadIdx.x;
  if (t < n) p[t] = v;
}
__global__ void fill_u32(unsigned* p, unsigned v, long long n) {
  long long t = blockIdx.x * (long long)blockDim.x + threadIdx.x;
  if (t < n) p[t] = v;
}

// ---------- xw = x[N,16] @ W[16,64]  (fp16 WMMA, K padded 16->32, fp32 accum) ----------
// W is staged transposed+f16 in LDS once per block; B-fragment = 1x ds_load_b128.
__global__ void gemm_xw(const float* __restrict__ x, const float* __restrict__ w,
                        float* __restrict__ xw, int nrows) {
  __shared__ __align__(16) _Float16 wt[HDIM * F_IN];   // WT[col][k]
  if (threadIdx.x < HDIM) {
    int col = threadIdx.x;
    #pragma unroll
    for (int k = 0; k < F_IN; ++k) wt[col * F_IN + k] = (_Float16)w[k * HDIM + col];
  }
  __syncthreads();

  int wid  = (blockIdx.x * blockDim.x + threadIdx.x) >> 5;
  int lane = threadIdx.x & 31;
  int hs = lane >> 4, lr = lane & 15;
  if (wid * 16 < nrows) {                        // uniform per wave; no return before barrier
    int row = wid * 16 + lr;

    // A 16x32 (16-bit) layout: lanes 0-15 hold K={0..7,16..23}, lanes 16-31 K={8..15,24..31}
    v16h a = {};                                 // real K=0..15; K>=16 stays zero
    const float4* xr4 = (const float4*)(x + row * F_IN + hs * 8);
    float4 p0 = xr4[0], p1 = xr4[1];
    a[0] = (_Float16)p0.x; a[1] = (_Float16)p0.y; a[2] = (_Float16)p0.z; a[3] = (_Float16)p0.w;
    a[4] = (_Float16)p1.x; a[5] = (_Float16)p1.y; a[6] = (_Float16)p1.z; a[7] = (_Float16)p1.w;

    #pragma unroll
    for (int nt = 0; nt < 4; ++nt) {
      // B 32x16: lanes 0-15 hold K=0..7, lanes 16-31 K=8..15 of column (nt*16+lr); K>=16 pad=0
      v16h b = {};
      {
        int col = nt * 16 + lr;
        const v8h* c8 = (const v8h*)(wt + col * F_IN + hs * 8);
        v8h lo = c8[0];
        #pragma unroll
        for (int j = 0; j < 8; ++j) { b[j] = lo[j]; }
      }
      v8f c = {};
      c = __builtin_amdgcn_wmma_f32_16x16x32_f16(false, a, false, b, (short)0, c, false, false);
      // C layout: VGPR r -> M = r + 8*hs, N = lr  (each (r,nt) is a 64B-coalesced segment)
      int col = nt * 16 + lr;
      int rb  = wid * 16 + hs * 8;
      #pragma unroll
      for (int r = 0; r < 8; ++r) xw[(rb + r) * HDIM + col] = c[r];
    }
  }
}

// ---------- generic out[M,64] = (optional BN(A))[M,64] @ W[64,64] + bias ----------
__global__ void gemm64(const float* __restrict__ A, const float* __restrict__ W,
                       const float* __restrict__ bias,
                       const float* __restrict__ scale, const float* __restrict__ shift,
                       float* __restrict__ out, int M) {
  __shared__ __align__(16) _Float16 wt[HDIM * HDIM];   // WT[col][k], 8KB
  for (int i = threadIdx.x; i < HDIM * HDIM; i += blockDim.x) {
    int k = i >> 6, col = i & 63;                      // coalesced global read
    wt[col * HDIM + k] = (_Float16)W[i];
  }
  __syncthreads();

  int wid  = (blockIdx.x * blockDim.x + threadIdx.x) >> 5;
  int lane = threadIdx.x & 31;
  int mt = wid >> 2, nt = wid & 3;
  int hs = lane >> 4, lr = lane & 15;
  if (mt * 16 < M) {
    int row = mt * 16 + lr;
    int col = nt * 16 + lr;

    v8f c = {};
    #pragma unroll
    for (int kb = 0; kb < HDIM; kb += 32) {
      v16h a = {}, b = {};
      // A: lanes 0-15 K = kb+{0..7,16..23}; lanes 16-31 K = kb+{8..15,24..31}
      const float4* a0 = (const float4*)(A + row * HDIM + kb + hs * 8);
      const float4* a1 = (const float4*)(A + row * HDIM + kb + 16 + hs * 8);
      float4 q0 = a0[0], q1 = a0[1], q2 = a1[0], q3 = a1[1];
      float f0[8] = { q0.x, q0.y, q0.z, q0.w, q1.x, q1.y, q1.z, q1.w };
      float f1[8] = { q2.x, q2.y, q2.z, q2.w, q3.x, q3.y, q3.z, q3.w };
      #pragma unroll
      for (int j = 0; j < 8; ++j) {
        int k0 = kb + hs * 8 + j, k1 = k0 + 16;
        float v0 = f0[j], v1 = f1[j];
        if (scale) { v0 = v0 * scale[k0] + shift[k0]; v1 = v1 * scale[k1] + shift[k1]; }
        a[j] = (_Float16)v0; a[8 + j] = (_Float16)v1;
      }
      // B: lanes 0-15 K = kb+{0..7,16..23}; lanes 16-31 K = kb+{8..15,24..31} of column col
      const v8h* c8 = (const v8h*)(wt + col * HDIM + kb + hs * 8);
      v8h lo = c8[0], hi = c8[2];
      #pragma unroll
      for (int j = 0; j < 8; ++j) { b[j] = lo[j]; b[8 + j] = hi[j]; }
      c = __builtin_amdgcn_wmma_f32_16x16x32_f16(false, a, false, b, (short)0, c, false, false);
    }
    int rb = mt * 16 + hs * 8;
    #pragma unroll
    for (int r = 0; r < 8; ++r) out[(rb + r) * HDIM + col] = c[r] + bias[col];
  }
}

// ---------- degree / norm ----------
__global__ void degree_count(const int* __restrict__ dst, float* __restrict__ deg, long long e) {
  long long t = blockIdx.x * (long long)blockDim.x + threadIdx.x;
  if (t < e) atomicAdd(&deg[dst[t]], 1.0f);
}
__global__ void deg_to_dinv(float* __restrict__ d, long long n) {
  long long t = blockIdx.x * (long long)blockDim.x + threadIdx.x;
  if (t < n) d[t] = rsqrtf(d[t] + 1.0f);
}

// ---------- h = xw*dinv^2 + bias (self-loop + bias term) ----------
__global__ void init_h(const float* __restrict__ xw, const float* __restrict__ dinv,
                       const float* __restrict__ bias, float* __restrict__ h, long long n64) {
  long long t = blockIdx.x * (long long)blockDim.x + threadIdx.x;
  if (t >= n64) return;
  long long row = t >> 6; int f = (int)(t & 63);
  float di = dinv[row];
  h[t] = xw[t] * di * di + bias[f];
}

// ---------- edge scatter: h[dst] += dinv[src]*dinv[dst] * xw[src]  (32 lanes/edge) ----------
__global__ void scatter_edges(const int* __restrict__ src, const int* __restrict__ dst,
                              const float* __restrict__ dinv, const float* __restrict__ xw,
                              float* __restrict__ h, long long ne) {
  long long t = blockIdx.x * (long long)blockDim.x + threadIdx.x;
  long long e = t >> 5;                 // uniform per wave
  if (e >= ne) return;
  int lane = threadIdx.x & 31;
  int s = src[e], d = dst[e];
  float norm = dinv[s] * dinv[d];
  const float2 v = ((const float2*)(xw + (long long)s * HDIM))[lane];
  float* hd = h + (long long)d * HDIM + lane * 2;
  atomicAdd(hd,     norm * v.x);
  atomicAdd(hd + 1, norm * v.y);
}

// ---------- per-column sum/sumsq over many rows (partials via atomics) ----------
__global__ void colstats_big(const float* __restrict__ A, float* __restrict__ sums, long long rows) {
  int f = threadIdx.x & 63;
  long long r0 = (long long)blockIdx.x * (blockDim.x >> 6) + (threadIdx.x >> 6);
  long long rstride = (long long)gridDim.x * (blockDim.x >> 6);
  float s = 0.f, s2 = 0.f;
  for (long long r = r0; r < rows; r += rstride) {
    float v = A[r * HDIM + f]; s += v; s2 += v * v;
  }
  atomicAdd(&sums[f], s);
  atomicAdd(&sums[64 + f], s2);
}
__global__ void finalize_bn(const float* __restrict__ sums, float invn,
                            const float* __restrict__ g, const float* __restrict__ bt,
                            float* __restrict__ scale, float* __restrict__ shift) {
  int c = threadIdx.x; if (c >= HDIM) return;
  float mean = sums[c] * invn;
  float var  = sums[64 + c] * invn - mean * mean;
  float sc = g[c] * rsqrtf(var + BN_EPS);
  scale[c] = sc; shift[c] = bt[c] - mean * sc;
}
// small M: one block, 64 threads, finalize inline
__global__ void colstats_small(const float* __restrict__ A, int rows,
                               const float* __restrict__ g, const float* __restrict__ bt,
                               float* __restrict__ scale, float* __restrict__ shift) {
  int c = threadIdx.x; if (c >= HDIM) return;
  float s = 0.f, s2 = 0.f;
  for (int r = 0; r < rows; ++r) { float v = A[r * HDIM + c]; s += v; s2 += v * v; }
  float mean = s / rows, var = s2 / rows - mean * mean;
  float sc = g[c] * rsqrtf(var + BN_EPS);
  scale[c] = sc; shift[c] = bt[c] - mean * sc;
}

// ---------- BN + PReLU + segment-max pool (ordered-uint atomicMax) ----------
__global__ void bn_prelu_pool(const float* __restrict__ h, const float* __restrict__ scale,
                              const float* __restrict__ shift, const float* __restrict__ alpha,
                              const int* __restrict__ batch, unsigned* __restrict__ pool,
                              long long n64) {
  long long t = blockIdx.x * (long long)blockDim.x + threadIdx.x;
  if (t >= n64) return;
  long long row = t >> 6; int f = (int)(t & 63);
  float v = h[t] * scale[f] + shift[f];
  v = (v >= 0.f) ? v : alpha[0] * v;
  int g = batch[row];
  if ((unsigned)g < N_GRAPHS) atomicMax(&pool[(size_t)g * HDIM + f], f2ord(v));
}
__global__ void pool_decode(const unsigned* __restrict__ in, float* __restrict__ out, long long n) {
  long long t = blockIdx.x * (long long)blockDim.x + threadIdx.x;
  if (t < n) out[t] = ord2f(in[t]);
}
__global__ void bn_prelu_ew(const float* __restrict__ in, const float* __restrict__ scale,
                            const float* __restrict__ shift, const float* __restrict__ alpha,
                            float* __restrict__ out, long long n64) {
  long long t = blockIdx.x * (long long)blockDim.x + threadIdx.x;
  if (t >= n64) return;
  int f = (int)(t & 63);
  float v = in[t] * scale[f] + shift[f];
  out[t] = (v >= 0.f) ? v : alpha[0] * v;
}

static inline unsigned blks(long long n, int b) { return (unsigned)((n + b - 1) / b); }

extern "C" void kernel_launch(void* const* d_in, const int* in_sizes, int n_in,
                              void* d_out, int out_size, void* d_ws, size_t ws_size,
                              hipStream_t stream) {
  const long long N  = in_sizes[0] / F_IN;      // 100000
  const long long E  = in_sizes[2] / 2;         // 1600000
  const long long NH = N * HDIM;
  const long long BH = (long long)N_GRAPHS * HDIM;

  // ---- workspace layout (floats) ----
  float* ws = (float*)d_ws;
  float* XW[2]   = { ws,            ws + NH };
  float* Hb[2]   = { ws + 2 * NH,   ws + 3 * NH };
  float* DINV[2] = { ws + 4 * NH,   ws + 4 * NH + N };
  size_t off = (size_t)(4 * NH + 2 * N);
  unsigned* POOL[2] = { (unsigned*)(ws + off), (unsigned*)(ws + off + BH) }; off += 2 * BH;
  float* P[2] = { ws + off, ws + off + BH }; off += 2 * BH;
  float* Q[2] = { ws + off, ws + off + BH }; off += 2 * BH;
  float* CAT  = ws + off;                     off += 2 * BH;
  float* SUMS = ws + off;                     off += 128;
  float* SC1[2] = { ws + off, ws + off + 128 };        off += 256;  // scale|shift bn1
  float* SC2[2] = { ws + off, ws + off + 128 };        off += 256;  // scale|shift bn2
  float* SCC    = ws + off;                            off += 128;  // scale|shift cat bn

  const unsigned ORD_NEG_INF = 0x007FFFFFu;   // f2ord(-inf)

  for (int b = 0; b < 2; ++b) {
    const float* x     = (const float*)d_in[b];
    const int*   ei    = (const int*)d_in[2 + b];
    const int*   src   = ei;
    const int*   dstv  = ei + E;
    const int*   batch = (const int*)d_in[4 + b];
    int base = 6 + b * 10;
    const float* w_conv = (const float*)d_in[base + 0];
    const float* b_conv = (const float*)d_in[base + 1];
    const float* w_lin  = (const float*)d_in[base + 2];
    const float* b_lin  = (const float*)d_in[base + 3];
    const float* bn1_g  = (const float*)d_in[base + 4];
    const float* bn1_b  = (const float*)d_in[base + 5];
    const float* bn2_g  = (const float*)d_in[base + 6];
    const float* bn2_b  = (const float*)d_in[base + 7];
    const float* a1     = (const float*)d_in[base + 8];
    const float* a2     = (const float*)d_in[base + 9];

    // 1) xw = x @ W_conv   (WMMA, W staged in LDS)
    long long tiles = N / 16;
    gemm_xw<<<blks(tiles * 32, 256), 256, 0, stream>>>(x, w_conv, XW[b], (int)N);

    // 2) deg -> dinv
    fill_f32<<<blks(N, 256), 256, 0, stream>>>(DINV[b], 0.f, N);
    degree_count<<<blks(E, 256), 256, 0, stream>>>(dstv, DINV[b], E);
    deg_to_dinv<<<blks(N, 256), 256, 0, stream>>>(DINV[b], N);

    // 3) h = xw*dinv^2 + bias; then scatter edges into h
    init_h<<<blks(NH, 256), 256, 0, stream>>>(XW[b], DINV[b], b_conv, Hb[b], NH);
    scatter_edges<<<blks(E * 32, 256), 256, 0, stream>>>(src, dstv, DINV[b], XW[b], Hb[b], E);

    // 4) BN1 stats
    fill_f32<<<1, 128, 0, stream>>>(SUMS, 0.f, 128);
    colstats_big<<<240, 256, 0, stream>>>(Hb[b], SUMS, N);
    finalize_bn<<<1, 64, 0, stream>>>(SUMS, 1.f / (float)N, bn1_g, bn1_b, SC1[b], SC1[b] + 64);

    // 5) BN1 + PReLU + global max pool
    fill_u32<<<blks(BH, 256), 256, 0, stream>>>(POOL[b], ORD_NEG_INF, BH);
    bn_prelu_pool<<<blks(NH, 256), 256, 0, stream>>>(Hb[b], SC1[b], SC1[b] + 64, a1, batch, POOL[b], NH);
    pool_decode<<<blks(BH, 256), 256, 0, stream>>>(POOL[b], P[b], BH);

    // 6) pooled linear (WMMA), BN2 + PReLU into CAT
    gemm64<<<blks((N_GRAPHS / 16) * 4 * 32, 256), 256, 0, stream>>>(
        P[b], w_lin, b_lin, nullptr, nullptr, Q[b], N_GRAPHS);
    colstats_small<<<1, 64, 0, stream>>>(Q[b], N_GRAPHS, bn2_g, bn2_b, SC2[b], SC2[b] + 64);
    bn_prelu_ew<<<blks(BH, 256), 256, 0, stream>>>(Q[b], SC2[b], SC2[b] + 64, a2, CAT + b * BH, BH);
  }

  // 7) concat BN + final linear (WMMA with fused input-BN) -> d_out [1024, 64]
  const float* cat_bn_g = (const float*)d_in[26];
  const float* cat_bn_b = (const float*)d_in[27];
  const float* cat_w    = (const float*)d_in[28];
  const float* cat_b    = (const float*)d_in[29];
  colstats_small<<<1, 64, 0, stream>>>(CAT, 2 * N_GRAPHS, cat_bn_g, cat_bn_b, SCC, SCC + 64);
  gemm64<<<blks((2 * N_GRAPHS / 16) * 4 * 32, 256), 256, 0, stream>>>(
      CAT, cat_w, cat_b, SCC, SCC + 64, (float*)d_out, 2 * N_GRAPHS);
}
